// HybridQuantumForecaster_69853348102870
// MI455X (gfx1250) — matrix operand order repaired and checked by reading, not compiled, not run on hardware
//
#include <hip/hip_runtime.h>

// HybridQuantumForecaster for MI455X (gfx1250, wave32, WMMA + TDM).
// Phase 1: GRU via v_wmma_f32_16x16x32_f16; h kept on-chip for all 512 steps;
//          x tiles staged into LDS by the Tensor Data Mover (double-buffered).
// Phase 2: 8-qubit statevector sim + MLP, one wave per batch element.

namespace {
constexpr int kB      = 2048;
constexpr int kT      = 512;
constexpr int kIn     = 32;
constexpr int kH      = 128;
constexpr int kHS     = 136;   // padded LDS row stride in halves (16B aligned rows)
constexpr int kNQ     = 8;
constexpr int kDim    = 256;
constexpr int kLayers = 2;
}

typedef _Float16     v16h   __attribute__((ext_vector_type(16)));
typedef _Float16     v8h    __attribute__((ext_vector_type(8)));
typedef float        v8f    __attribute__((ext_vector_type(8)));
typedef float        v4f    __attribute__((ext_vector_type(4)));
typedef unsigned int ui32x4 __attribute__((ext_vector_type(4)));
typedef int          i32x4  __attribute__((ext_vector_type(4)));
typedef int          i32x8  __attribute__((ext_vector_type(8)));

// ---- fast activations: single v_exp_f32 / v_rcp_f32, no IEEE div sequence ----
static __device__ inline float fast_rcp(float x)     { return __builtin_amdgcn_rcpf(x); }
static __device__ inline float fast_sigmoid(float x) { return fast_rcp(1.f + __expf(-x)); }
static __device__ inline float fast_tanh(float x) {
  float e = __expf(-2.f * x);
  return (1.f - e) * fast_rcp(1.f + e);
}

// Load 16 consecutive f32 and convert to an f16 WMMA B-fragment half-row.
static __device__ inline v16h load_w16f(const float* __restrict__ p) {
  v4f a = *(const v4f*)(p + 0);
  v4f b = *(const v4f*)(p + 4);
  v4f c = *(const v4f*)(p + 8);
  v4f d = *(const v4f*)(p + 12);
  v16h r;
#pragma unroll
  for (int i = 0; i < 4; ++i) {
    r[i]      = (_Float16)a[i];
    r[4 + i]  = (_Float16)b[i];
    r[8 + i]  = (_Float16)c[i];
    r[12 + i] = (_Float16)d[i];
  }
  return r;
}

static __device__ inline v8f wmma_f16(v16h a, v16h b, v8f c) {
  // D = A(16x32 f16) x B(32x16 f16) + C(16x16 f32)
  return __builtin_amdgcn_wmma_f32_16x16x32_f16(false, a, false, b, (short)0, c,
                                                false, false);
}

// ---- Tensor Data Mover: DMA one 16-row x 32-f32 x-tile (row stride T*IN) ----
// D# per CDNA5 ISA ch.8: group0 = {count/flags, lds_addr, global_addr, type=2},
// group1 = {data_size=4B, tensor_dim0=32, tensor_dim1=16, tile 32x16,
//           dim0_stride = kT*kIn}.  Groups 2/3 unused (2D tensor).
static __device__ inline void tdm_load_x_tile(const float* gsrc, float* ldst) {
  unsigned long long ga = (unsigned long long)(uintptr_t)gsrc;
  unsigned lo = (unsigned)(uintptr_t)ldst;  // low 32 bits of flat ptr = LDS offset
  ui32x4 g0;
  g0[0] = 1u;                                 // count=1, user mode, no gather
  g0[1] = lo;                                 // lds_addr
  g0[2] = (unsigned)ga;                       // global_addr[31:0]
  g0[3] = (unsigned)(ga >> 32) | (2u << 30);  // global_addr[56:32] | type=2
  i32x8 g1;
  g1[0] = (int)(2u << 16);                    // wg_mask=0, data_size=2 (4 bytes)
  g1[1] = (int)((unsigned)kIn << 16);         // tensor_dim0[15:0]=32
  g1[2] = (int)(16u << 16);                   // tensor_dim0 hi=0 | tensor_dim1 lo=16
  g1[3] = (int)((unsigned)kIn << 16);         // tensor_dim1 hi=0 | tile_dim0=32
  g1[4] = 16;                                 // tile_dim1=16, tile_dim2=0
  g1[5] = kT * kIn;                           // tensor_dim0_stride lo32 = 16384
  g1[6] = 0;                                  // stride hi | dim1_stride lo (unused)
  g1[7] = 0;
  i32x4 z4 = {0, 0, 0, 0};
#if defined(__clang_major__) && __clang_major__ >= 23
  i32x8 z8 = {0, 0, 0, 0, 0, 0, 0, 0};
  __builtin_amdgcn_tensor_load_to_lds(g0, g1, z4, z4, z8, 0);
#else
  __builtin_amdgcn_tensor_load_to_lds(g0, g1, z4, z4, 0);
#endif
}

// ---------------------------------------------------------------------------
// Kernel 1: GRU. Grid = B/16 workgroups, 256 threads (8 waves).
// Wave w owns gate columns [16w,16w+16) of r, z and n -> gate math stays in
// registers in WMMA C-layout.  h (16x128 f16 A-operand) and the x tile are
// double-buffered in LDS -> one barrier per timestep.
// ---------------------------------------------------------------------------
__global__ __launch_bounds__(256) void gru_wmma_kernel(
    const float* __restrict__ x,   const float* __restrict__ Wih,
    const float* __restrict__ Whh, const float* __restrict__ bih,
    const float* __restrict__ bhh, float* __restrict__ hout)
{
  __shared__ alignas(16) _Float16 hbuf[2][16 * kHS];
  __shared__ alignas(16) float    xstage[2][16 * kIn];

  const int  tid  = threadIdx.x;
  const int  wave = tid >> 5;
  const int  lane = tid & 31;
  const int  ll   = lane & 15;
  const bool hi   = lane >= 16;
  const int  b0   = blockIdx.x * 16;

  // Gate-column bases for this wave's r / z / n tiles.
  const int nR = 16 * wave;
  const int nZ = kH + 16 * wave;
  const int nN = 2 * kH + 16 * wave;

  // B-fragments: lanes 0-15 hold K=kb..kb+15 for column N=ll, lanes 16-31 the
  // upper K half.  W rows are the gate columns (W.T is the K x N operand).
  const int kb = hi ? 16 : 0;
  v16h BxR = load_w16f(Wih + (size_t)(nR + ll) * kIn + kb);
  v16h BxZ = load_w16f(Wih + (size_t)(nZ + ll) * kIn + kb);
  v16h BxN = load_w16f(Wih + (size_t)(nN + ll) * kIn + kb);
  v16h BhR[4], BhZ[4], BhN[4];
#pragma unroll
  for (int kk = 0; kk < 4; ++kk) {
    BhR[kk] = load_w16f(Whh + (size_t)(nR + ll) * kH + 32 * kk + kb);
    BhZ[kk] = load_w16f(Whh + (size_t)(nZ + ll) * kH + 32 * kk + kb);
    BhN[kk] = load_w16f(Whh + (size_t)(nN + ll) * kH + 32 * kk + kb);
  }
  const float bcR = bih[nR + ll] + bhh[nR + ll];
  const float bcZ = bih[nZ + ll] + bhh[nZ + ll];
  const float bNX = bih[nN + ll];
  const float bNH = bhh[nN + ll];

  // Prologue: zero h buffer 0, stage x(t=0) tile with plain loads.
  for (int i = tid; i < 16 * kHS; i += 256) hbuf[0][i] = (_Float16)0.f;
  for (int i = tid; i < 16 * kIn; i += 256)
    xstage[0][i] = x[((size_t)(b0 + (i >> 5)) * kT) * kIn + (i & 31)];
  __syncthreads();

  v8f hreg = {};               // h in C-layout: elem i -> row i (+8 hi lanes)
  const int cb = hi ? 16 : 0;  // A-layout K offset for this lane half
  const int m0 = hi ? 8 : 0;

  auto step = [&](int t, const _Float16* __restrict__ hrd,
                  _Float16* __restrict__ hwr,
                  const float* __restrict__ xcur, float* __restrict__ xnxt) {
    // Kick off DMA of the next x tile while this step computes.
    if (wave == 0 && t + 1 < kT)
      tdm_load_x_tile(x + ((size_t)b0 * kT + (t + 1)) * kIn, xnxt);

    // x A-fragment (16x32) from LDS stage: K = cb..cb+15.
    const float* xs = xcur + ll * kIn + cb;
    v4f x0 = *(const v4f*)(xs + 0);
    v4f x1 = *(const v4f*)(xs + 4);
    v4f x2 = *(const v4f*)(xs + 8);
    v4f x3 = *(const v4f*)(xs + 12);
    v16h ax;
#pragma unroll
    for (int i = 0; i < 4; ++i) {
      ax[i]      = (_Float16)x0[i];
      ax[4 + i]  = (_Float16)x1[i];
      ax[8 + i]  = (_Float16)x2[i];
      ax[12 + i] = (_Float16)x3[i];
    }

    v8f accR = {}, accZ = {}, accNX = {}, accNH = {};
    accR  = wmma_f16(ax, BxR, accR);     // i_r
    accZ  = wmma_f16(ax, BxZ, accZ);     // i_z
    accNX = wmma_f16(ax, BxN, accNX);    // i_n (kept separate from h_n)

#pragma unroll
    for (int kk = 0; kk < 4; ++kk) {     // K = 128 of h @ W_hh.T
      const int c0 = 32 * kk + cb;
      v8h h0 = *(const v8h*)&hrd[ll * kHS + c0];
      v8h h1 = *(const v8h*)&hrd[ll * kHS + c0 + 8];
      v16h ah;
#pragma unroll
      for (int i = 0; i < 8; ++i) { ah[i] = h0[i]; ah[8 + i] = h1[i]; }
      accR  = wmma_f16(ah, BhR[kk], accR);
      accZ  = wmma_f16(ah, BhZ[kk], accZ);
      accNH = wmma_f16(ah, BhN[kk], accNH);
    }

    // Gate math entirely in registers (C-layout rows x this wave's columns).
#pragma unroll
    for (int i = 0; i < 8; ++i) {
      float r = fast_sigmoid(accR[i] + bcR);
      float z = fast_sigmoid(accZ[i] + bcZ);
      float n = fast_tanh(accNX[i] + bNX + r * (accNH[i] + bNH));
      hreg[i] = (1.f - z) * n + z * hreg[i];
    }
#pragma unroll
    for (int i = 0; i < 8; ++i)
      hwr[(i + m0) * kHS + 16 * wave + ll] = (_Float16)hreg[i];

    // Make TDM'd x tile + updated h visible to all waves: wait then barrier.
    if (wave == 0 && t + 1 < kT) __builtin_amdgcn_s_wait_tensorcnt(0);
    __syncthreads();
  };

  for (int t = 0; t < kT; t += 2) {
    step(t,     hbuf[0], hbuf[1], xstage[0], xstage[1]);
    step(t + 1, hbuf[1], hbuf[0], xstage[1], xstage[0]);
  }

#pragma unroll
  for (int i = 0; i < 8; ++i)
    hout[(size_t)(b0 + i + m0) * kH + 16 * wave + ll] = hreg[i];
}

// ---------------------------------------------------------------------------
// Kernel 2: q_in -> 8-qubit statevector -> signed readout -> 8->16->2 MLP.
// Grid = B/8 workgroups, 256 threads; one wave per batch element, state in LDS.
// Index bit for wire w is bit (7-w) of the flattened amplitude index.
// ---------------------------------------------------------------------------
__global__ __launch_bounds__(256) void qsim_mlp_kernel(
    const float* __restrict__ hin, const float* __restrict__ Wq,
    const float* __restrict__ bq,  const float* __restrict__ qw,
    const float* __restrict__ W1,  const float* __restrict__ b1,
    const float* __restrict__ W2,  const float* __restrict__ b2,
    float* __restrict__ out)
{
  __shared__ float sre[8][kDim];
  __shared__ float sim_[8][kDim];
  __shared__ float camp[8][kNQ];
  __shared__ float samp[8][kNQ];

  const int wave = threadIdx.x >> 5;
  const int lane = threadIdx.x & 31;
  const int b    = blockIdx.x * 8 + wave;

  if (lane < kNQ) {
    const float* hp = hin + (size_t)b * kH;
    const float* wp = Wq + lane * kH;
    float d = bq[lane];
    for (int c = 0; c < kH; ++c) d = fmaf(hp[c], wp[c], d);
    float half = fast_tanh(d) * 0.78539816339744831f;  // (pi/2) * 0.5
    camp[wave][lane] = __cosf(half);
    samp[wave][lane] = __sinf(half);
  }
  __syncthreads();

  // Product initial state (real).
#pragma unroll
  for (int e = 0; e < 8; ++e) {
    int idx = lane * 8 + e;
    float re = 1.f;
#pragma unroll
    for (int k = 0; k < kNQ; ++k)
      re *= ((idx >> (7 - k)) & 1) ? samp[wave][k] : camp[wave][k];
    sre[wave][idx]  = re;
    sim_[wave][idx] = 0.f;
  }
  __syncthreads();

  for (int l = 0; l < kLayers; ++l) {
    // RX(theta) on each wire: [[c,-is],[-is,c]]
    for (int w = 0; w < kNQ; ++w) {
      const int   p  = 7 - w;
      const float th = qw[l * kNQ + w] * 0.5f;
      const float c  = __cosf(th), s = __sinf(th);
#pragma unroll
      for (int e = 0; e < 8; ++e) {
        int i0 = lane * 8 + e;
        if (((i0 >> p) & 1) == 0) {
          int i1 = i0 | (1 << p);
          float ar = sre[wave][i0], ai = sim_[wave][i0];
          float br = sre[wave][i1], bi = sim_[wave][i1];
          sre[wave][i0]  = c * ar + s * bi;
          sim_[wave][i0] = c * ai - s * br;
          sre[wave][i1]  = c * br + s * ai;
          sim_[wave][i1] = c * bi - s * ar;
        }
      }
      __syncthreads();
    }
    // CNOT ring: control g, target (g+1)%8 — swap target bit where control=1.
    for (int g = 0; g < kNQ; ++g) {
      const int pc = 7 - g;
      const int pt = 7 - ((g + 1) & 7);
#pragma unroll
      for (int e = 0; e < 8; ++e) {
        int i0 = lane * 8 + e;
        if (((i0 >> pc) & 1) == 1 && ((i0 >> pt) & 1) == 0) {
          int i1 = i0 | (1 << pt);
          float tr = sre[wave][i0];  sre[wave][i0]  = sre[wave][i1];  sre[wave][i1]  = tr;
          float ti = sim_[wave][i0]; sim_[wave][i0] = sim_[wave][i1]; sim_[wave][i1] = ti;
        }
      }
      __syncthreads();
    }
  }

  // q_out[k] = sum_idx |amp|^2 * (1 - 2*bit_{7-k}(idx)); wave shuffle reduce.
  float acc[kNQ];
#pragma unroll
  for (int k = 0; k < kNQ; ++k) acc[k] = 0.f;
#pragma unroll
  for (int e = 0; e < 8; ++e) {
    int idx = lane * 8 + e;
    float re = sre[wave][idx], im = sim_[wave][idx];
    float p2 = re * re + im * im;
#pragma unroll
    for (int k = 0; k < kNQ; ++k)
      acc[k] += ((idx >> (7 - k)) & 1) ? -p2 : p2;
  }
#pragma unroll
  for (int off = 16; off >= 1; off >>= 1)
#pragma unroll
    for (int k = 0; k < kNQ; ++k)
      acc[k] += __shfl_xor(acc[k], off, 32);

  if (lane == 0) {
    float o0 = b2[0], o1 = b2[1];
#pragma unroll
    for (int j = 0; j < 16; ++j) {
      float v = b1[j];
#pragma unroll
      for (int k = 0; k < kNQ; ++k) v = fmaf(acc[k], W1[j * kNQ + k], v);
      v = fmaxf(v, 0.f);
      o0 = fmaf(v, W2[j], o0);
      o1 = fmaf(v, W2[16 + j], o1);
    }
    out[b * 2 + 0] = o0;
    out[b * 2 + 1] = o1;
  }
}

extern "C" void kernel_launch(void* const* d_in, const int* in_sizes, int n_in,
                              void* d_out, int out_size, void* d_ws, size_t ws_size,
                              hipStream_t stream) {
  (void)in_sizes; (void)n_in; (void)out_size; (void)ws_size;
  const float* x   = (const float*)d_in[0];
  const float* Wih = (const float*)d_in[1];
  const float* Whh = (const float*)d_in[2];
  const float* bih = (const float*)d_in[3];
  const float* bhh = (const float*)d_in[4];
  const float* Wq  = (const float*)d_in[5];
  const float* bq  = (const float*)d_in[6];
  const float* qw  = (const float*)d_in[7];
  const float* W1  = (const float*)d_in[8];
  const float* b1  = (const float*)d_in[9];
  const float* W2  = (const float*)d_in[10];
  const float* b2  = (const float*)d_in[11];

  float* hws = (float*)d_ws;  // h_last: 2048 x 128 f32 = 1 MiB scratch

  gru_wmma_kernel<<<kB / 16, 256, 0, stream>>>(x, Wih, Whh, bih, bhh, hws);
  qsim_mlp_kernel<<<kB / 8, 256, 0, stream>>>(hws, Wq, bq, qw, W1, b1, W2, b2,
                                              (float*)d_out);
}
